// Mlp_FMoE_40132174414263
// MI455X (gfx1250) — matrix operand order repaired and checked by reading, compiled, not run on hardware
//
#include <hip/hip_runtime.h>
#include <hip/hip_bf16.h>

typedef __bf16 bf16_t;
typedef __attribute__((ext_vector_type(16))) __bf16 v16bf;
typedef __attribute__((ext_vector_type(8)))  __bf16 v8bf;
typedef __attribute__((ext_vector_type(8)))  float   v8f;
typedef __attribute__((ext_vector_type(4)))  unsigned int u32x4;
typedef __attribute__((ext_vector_type(8)))  int i32x8;
typedef __attribute__((ext_vector_type(4)))  int i32x4;

// ---------------------------------------------------------------------------
// Tensor Data Mover: async 2D tile load Global -> LDS (CDNA5 TDM, D# per
// CDNA5 ISA). Tile = tile_h rows x tile_w elems (2B each),
// row stride = row_stride elems. pad_interval=16 dwords / pad_amount=4 dwords
// makes the TDM itself emit the 80B-padded LDS row stride (64B row + 16B pad).
// ---------------------------------------------------------------------------
__device__ __forceinline__ void tdm_load_2d_bf16(
    unsigned lds_addr, const bf16_t* gptr, unsigned tensor_w, unsigned tensor_h,
    unsigned tile_w, unsigned tile_h, unsigned row_stride) {
  unsigned long long ga = (unsigned long long)(uintptr_t)gptr;
  u32x4 g0;
  g0[0] = 1u;                                   // count=1, user descriptor
  g0[1] = lds_addr;                             // LDS byte address (low 32 of generic)
  g0[2] = (unsigned)(ga & 0xffffffffu);         // global_addr[31:0]
  g0[3] = (unsigned)((ga >> 32) & 0x01ffffffu)  // global_addr[56:32]
          | (2u << 30);                         // type=2 ("image")
  i32x8 g1;
  g1[0] = (int)((1u << 16)      // data_size=1 -> 2 bytes
                | (1u << 20)    // pad_enable
                | (3u << 22)    // pad_interval: 16 dwords (64B = one tile row)
                | (3u << 25));  // pad_amount: 4 dwords (16B)
  g1[1] = (int)((tensor_w & 0xffffu) << 16);                       // dim0 lo
  g1[2] = (int)((tensor_w >> 16) | ((tensor_h & 0xffffu) << 16));  // dim0 hi | dim1 lo
  g1[3] = (int)((tensor_h >> 16) | (tile_w << 16));                // dim1 hi | tile_dim0
  g1[4] = (int)tile_h;                                             // tile_dim1 (tile_dim2=0)
  g1[5] = (int)row_stride;                                         // dim0_stride[31:0]
  g1[6] = 0;                                                       // stride hi bits
  g1[7] = 0;
  i32x4 gz = {0, 0, 0, 0};  // groups 2/3 unused (2D tensor)
#if __clang_major__ >= 23
  i32x8 gz8 = {0, 0, 0, 0, 0, 0, 0, 0};
  __builtin_amdgcn_tensor_load_to_lds(g0, g1, gz, gz, gz8, 0);
#else
  __builtin_amdgcn_tensor_load_to_lds(g0, g1, gz, gz, 0);
#endif
}

// ---------------------------------------------------------------------------
// Elementwise converts / shift-quant
// ---------------------------------------------------------------------------
__global__ __launch_bounds__(256) void f32_to_bf16_kernel(
    const float* __restrict__ in, bf16_t* __restrict__ out, long n) {
  long i = (long)blockIdx.x * blockDim.x + threadIdx.x;
  if (i < n) out[i] = (bf16_t)in[i];
}

__global__ __launch_bounds__(256) void shiftquant_to_bf16_kernel(
    const float* __restrict__ in, bf16_t* __restrict__ out, long n) {
  long i = (long)blockIdx.x * blockDim.x + threadIdx.x;
  if (i < n) {
    float w = in[i];
    float s = (w > 0.0f) ? 1.0f : ((w < 0.0f) ? -1.0f : 0.0f);
    float sh = rintf(log2f(fabsf(w) + 1e-12f));      // RNE == jnp.round
    sh = fminf(fmaxf(sh, -14.0f), 0.0f);             // deepshift clip
    out[i] = (bf16_t)(s * exp2f(sh));                // exact in bf16
  }
}

// ---------------------------------------------------------------------------
// Top-1 gate: one wave per token, dual dot product, wave32 butterfly reduce
// ---------------------------------------------------------------------------
__global__ __launch_bounds__(256) void gate_f32_kernel(
    const float* __restrict__ X, const float* __restrict__ Wg,
    unsigned char* __restrict__ pick, int T, int K) {
  int wave = threadIdx.x >> 5, lane = threadIdx.x & 31;
  int t = blockIdx.x * 8 + wave;
  if (t >= T) return;
  float l0 = 0.f, l1 = 0.f;
  for (int c = lane; c < K; c += 32) {
    float xv = X[(size_t)t * K + c];
    l0 += xv * Wg[c * 2 + 0];
    l1 += xv * Wg[c * 2 + 1];
  }
  for (int off = 16; off > 0; off >>= 1) {
    l0 += __shfl_xor(l0, off, 32);
    l1 += __shfl_xor(l1, off, 32);
  }
  if (lane == 0) pick[t] = (l0 >= l1) ? 1 : 0;   // argmax tie -> expert 0
}

__global__ __launch_bounds__(256) void gate_bf16_kernel(
    const bf16_t* __restrict__ X, const float* __restrict__ Wg,
    unsigned char* __restrict__ pick, int T, int K) {
  int wave = threadIdx.x >> 5, lane = threadIdx.x & 31;
  int t = blockIdx.x * 8 + wave;
  if (t >= T) return;
  float l0 = 0.f, l1 = 0.f;
  for (int c = lane; c < K; c += 32) {
    float xv = (float)X[(size_t)t * K + c];
    l0 += xv * Wg[c * 2 + 0];
    l1 += xv * Wg[c * 2 + 1];
  }
  for (int off = 16; off > 0; off >>= 1) {
    l0 += __shfl_xor(l0, off, 32);
    l1 += __shfl_xor(l1, off, 32);
  }
  if (lane == 0) pick[t] = (l0 >= l1) ? 1 : 0;
}

// ---------------------------------------------------------------------------
// Dual-expert MoE GEMM: Y[t,n] = pick0[t] ? (A@W0^T + b0) : (A@W1^T + b1)
// A: [T,K] bf16 row-major, W0/W1: [N,K] bf16 row-major (torch [out,in]).
// Block = 256 threads (8 waves): 256 tokens x 64 outputs (32 rows per wave).
// Weight slabs (2 experts x 64 rows x 32 halfs) are staged by the Tensor Data
// Mover: wave 0 DMAs expert 0, wave 1 expert 1, double-buffered in padded LDS
// (TDM pad fields emit the 80B row stride). Per K-step per wave: 16x
// v_wmma_f32_16x16x32_bf16; each B fragment (2x ds_load_b128) feeds 2 WMMAs.
// ---------------------------------------------------------------------------
template <bool OUT_BF16>
__global__ __launch_bounds__(256) void moe_gemm_bf16_kernel(
    const bf16_t* __restrict__ A, const bf16_t* __restrict__ W0,
    const bf16_t* __restrict__ W1, const float* __restrict__ bias0,
    const float* __restrict__ bias1, const unsigned char* __restrict__ pick0,
    void* __restrict__ out, int T, int K, int Nout) {
  constexpr int LDSROW = 40;  // 32 halfs + 8 pad -> 80B stride (TDM-generated)
  __shared__ __attribute__((aligned(32))) bf16_t sB[2][2][64][LDSROW];

  const int wave = threadIdx.x >> 5;
  const int lane = threadIdx.x & 31;
  const int laneHalf = lane >> 4;   // K-half select per ISA A/B layouts
  const int laneMod  = lane & 15;   // row (A) / col (B) within tile

  const int m0 = blockIdx.x * 256 + wave * 32;
  const int n0 = blockIdx.y * 64;

  // A rows for this lane, both M tiles (clamped; no divergent return: WMMA
  // requires EXEC all-1s)
  int ar0 = m0 + laneMod;      if (ar0 >= T) ar0 = T - 1;
  int ar1 = m0 + 16 + laneMod; if (ar1 >= T) ar1 = T - 1;
  const bf16_t* aBase0 = A + (size_t)ar0 * K;
  const bf16_t* aBase1 = A + (size_t)ar1 * K;

  // TDM staging: wave w (w<2) owns expert w's 64x32 slab
  const bf16_t* wTile = (wave == 1 ? W1 : W0) + (size_t)n0 * K;
  auto stage = [&](int buf, int k0) {
    tdm_load_2d_bf16((unsigned)(uintptr_t)&sB[buf][wave][0][0], wTile + k0,
                     (unsigned)K, (unsigned)Nout, 32u, 64u, (unsigned)K);
  };

  union BV { v16bf v; v8bf h[2]; };
  auto loadB = [&](int buf, int ex, int s) {
    BV b;
    const bf16_t* bp = &sB[buf][ex][s * 16 + laneMod][laneHalf * 8];
    b.h[0] = *(const v8bf*)(bp + 0);
    b.h[1] = *(const v8bf*)(bp + 16);
    return b;
  };

  v8f acc[2][2][4] = {};  // [expert][m-tile][n-subtile]

  // prologue: DMA first slab into buffer 0
  if (wave < 2) stage(0, 0);

  for (int k0 = 0, it = 0; k0 < K; k0 += 32, ++it) {
    const int buf = it & 1;
    const bool hasNext = (k0 + 32) < K;

    if (wave < 2) __builtin_amdgcn_s_wait_tensorcnt(0);  // slab `buf` landed
    __syncthreads();  // publish slab; all reads of `buf^1` retired

    // DMA next slab into the other buffer; overlaps this step's WMMAs
    if (hasNext && wave < 2) stage(buf ^ 1, k0 + 32);

    // A fragments per ISA 16-bit 16x32 layout: lane half h holds
    // K = [8h, 8h+8) and [16+8h, 16+8h+8)
    BV a0, a1;
    a0.h[0] = *(const v8bf*)(aBase0 + k0 + laneHalf * 8);
    a0.h[1] = *(const v8bf*)(aBase0 + k0 + 16 + laneHalf * 8);
    a1.h[0] = *(const v8bf*)(aBase1 + k0 + laneHalf * 8);
    a1.h[1] = *(const v8bf*)(aBase1 + k0 + 16 + laneHalf * 8);
    if (hasNext) {
      __builtin_prefetch(aBase0 + k0 + 32, 0, 1);  // global_prefetch_b8
      __builtin_prefetch(aBase1 + k0 + 32, 0, 1);
    }

    // B pipeline: each fragment feeds 2 WMMAs (both M tiles)
    BV bcur = loadB(buf, 0, 0);
#pragma unroll
    for (int i = 0; i < 8; ++i) {
      const int ex = i >> 2, s = i & 3;
      BV bnext;
      if (i < 7) bnext = loadB(buf, (i + 1) >> 2, (i + 1) & 3);
      acc[ex][0][s] = __builtin_amdgcn_wmma_f32_16x16x32_bf16(
          false, a0.v, false, bcur.v, (short)0, acc[ex][0][s], false, false);
      acc[ex][1][s] = __builtin_amdgcn_wmma_f32_16x16x32_bf16(
          false, a1.v, false, bcur.v, (short)0, acc[ex][1][s], false, false);
      bcur = bnext;
    }
  }

  // Epilogue: per-row expert select + bias. C/D layout: VGPR r, lanes 0-15 ->
  // M=r, lanes 16-31 -> M=r+8; N = lane&15.
  bool pk[2][8];
#pragma unroll
  for (int mt = 0; mt < 2; ++mt)
#pragma unroll
    for (int r = 0; r < 8; ++r) {
      int row = m0 + mt * 16 + laneHalf * 8 + r;
      pk[mt][r] = pick0[row < T ? row : T - 1] != 0;
    }
#pragma unroll
  for (int s = 0; s < 4; ++s) {
    const int col = n0 + s * 16 + laneMod;
    if (col >= Nout) continue;
    const float b0c = bias0[col];
    const float b1c = bias1[col];
#pragma unroll
    for (int mt = 0; mt < 2; ++mt) {
#pragma unroll
      for (int r = 0; r < 8; ++r) {
        const int row = m0 + mt * 16 + laneHalf * 8 + r;
        if (row >= T) continue;
        float v = pk[mt][r] ? (acc[0][mt][s][r] + b0c)
                            : (acc[1][mt][s][r] + b1c);
        if (OUT_BF16)
          ((bf16_t*)out)[(size_t)row * Nout + col] = (bf16_t)v;
        else
          ((float*)out)[(size_t)row * Nout + col] = v;
      }
    }
  }
}

// ---------------------------------------------------------------------------
// Depthwise 3x3 SAME conv + bias + exact GELU. h layout [B, H*W, Dh],
// channel-fastest => fully coalesced. H/W read from device scalars.
// ---------------------------------------------------------------------------
__global__ __launch_bounds__(256) void dwconv_gelu_kernel(
    const bf16_t* __restrict__ Hin, const float* __restrict__ Wc,
    const float* __restrict__ bc, const int* __restrict__ Hp,
    const int* __restrict__ Wp, bf16_t* __restrict__ Hout, int T, int Dh) {
  long idx = (long)blockIdx.x * blockDim.x + threadIdx.x;
  if (idx >= (long)T * Dh) return;
  int d = (int)(idx % Dh);
  int t = (int)(idx / Dh);
  int Hh = *Hp, Ww = *Wp;
  int N = Hh * Ww;
  int b = t / N, n = t % N;
  int y = n / Ww, x = n % Ww;
  float acc = 0.0f;
#pragma unroll
  for (int dy = -1; dy <= 1; ++dy) {
    int yy = y + dy;
    if (yy < 0 || yy >= Hh) continue;
#pragma unroll
    for (int dx = -1; dx <= 1; ++dx) {
      int xx = x + dx;
      if (xx < 0 || xx >= Ww) continue;
      int tt = b * N + yy * Ww + xx;
      acc += (float)Hin[(size_t)tt * Dh + d] * Wc[d * 9 + (dy + 1) * 3 + (dx + 1)];
    }
  }
  acc += bc[d];
  float g = 0.5f * acc * (1.0f + erff(acc * 0.70710678118654752f));  // exact GELU
  Hout[idx] = (bf16_t)g;
}

// ---------------------------------------------------------------------------
// Launch
// ---------------------------------------------------------------------------
extern "C" void kernel_launch(void* const* d_in, const int* in_sizes, int n_in,
                              void* d_out, int out_size, void* d_ws, size_t ws_size,
                              hipStream_t stream) {
  const float* x      = (const float*)d_in[0];
  const int*   Hp     = (const int*)d_in[1];
  const int*   Wp     = (const int*)d_in[2];
  const float* wg1    = (const float*)d_in[3];
  const float* fc1_w0 = (const float*)d_in[4];
  const float* fc1_b0 = (const float*)d_in[5];
  const float* fc1_w1 = (const float*)d_in[6];
  const float* fc1_b1 = (const float*)d_in[7];
  const float* dw_w   = (const float*)d_in[8];
  const float* dw_b   = (const float*)d_in[9];
  const float* wg2    = (const float*)d_in[10];
  const float* fc2_w0 = (const float*)d_in[11];
  const float* fc2_b0 = (const float*)d_in[12];
  const float* fc2_w1 = (const float*)d_in[13];
  const float* fc2_b1 = (const float*)d_in[14];

  const int C  = in_sizes[3] / 2;   // wg1: [C,2]
  const int Dh = in_sizes[5];       // fc1_b0: [Dh]
  const int T  = in_sizes[0] / C;   // x: [B,N,C]

  // workspace layout (256B aligned chunks)
  char* p = (char*)d_ws;
  auto alloc = [&](size_t bytes) -> char* {
    char* r = p;
    p += (bytes + 255) & ~(size_t)255;
    return r;
  };
  bf16_t* xb    = (bf16_t*)alloc((size_t)T * C * sizeof(bf16_t));
  bf16_t* w1_0b = (bf16_t*)alloc((size_t)Dh * C * sizeof(bf16_t));
  bf16_t* w1_1b = (bf16_t*)alloc((size_t)Dh * C * sizeof(bf16_t));
  bf16_t* w2_0b = (bf16_t*)alloc((size_t)C * Dh * sizeof(bf16_t));
  bf16_t* w2_1b = (bf16_t*)alloc((size_t)C * Dh * sizeof(bf16_t));
  bf16_t* hbuf  = (bf16_t*)alloc((size_t)T * Dh * sizeof(bf16_t));
  bf16_t* hgbuf = (bf16_t*)alloc((size_t)T * Dh * sizeof(bf16_t));
  unsigned char* pick1 = (unsigned char*)alloc((size_t)T);
  unsigned char* pick2 = (unsigned char*)alloc((size_t)T);

  const int BT = 256;
  auto blk = [&](long n) { return (unsigned)((n + BT - 1) / BT); };

  // 1) precision conversion + shift quantization
  f32_to_bf16_kernel<<<blk((long)T * C), BT, 0, stream>>>(x, xb, (long)T * C);
  f32_to_bf16_kernel<<<blk((long)Dh * C), BT, 0, stream>>>(fc1_w0, w1_0b, (long)Dh * C);
  shiftquant_to_bf16_kernel<<<blk((long)Dh * C), BT, 0, stream>>>(fc1_w1, w1_1b, (long)Dh * C);
  f32_to_bf16_kernel<<<blk((long)C * Dh), BT, 0, stream>>>(fc2_w0, w2_0b, (long)C * Dh);
  shiftquant_to_bf16_kernel<<<blk((long)C * Dh), BT, 0, stream>>>(fc2_w1, w2_1b, (long)C * Dh);

  // 2) gate 1 (fp32 activations)
  gate_f32_kernel<<<(T + 7) / 8, BT, 0, stream>>>(x, wg1, pick1, T, C);

  // 3) fc1 MoE GEMM -> h (bf16)
  {
    dim3 grid((T + 255) / 256, (Dh + 63) / 64);
    moe_gemm_bf16_kernel<true><<<grid, BT, 0, stream>>>(
        xb, w1_0b, w1_1b, fc1_b0, fc1_b1, pick1, (void*)hbuf, T, C, Dh);
  }

  // 4) depthwise conv + bias + GELU -> hg (bf16)
  dwconv_gelu_kernel<<<blk((long)T * Dh), BT, 0, stream>>>(
      hbuf, dw_w, dw_b, Hp, Wp, hgbuf, T, Dh);

  // 5) gate 2 (bf16 activations)
  gate_bf16_kernel<<<(T + 7) / 8, BT, 0, stream>>>(hgbuf, wg2, pick2, T, Dh);

  // 6) fc2 MoE GEMM -> d_out (f32)
  {
    dim3 grid((T + 255) / 256, (C + 63) / 64);
    moe_gemm_bf16_kernel<false><<<grid, BT, 0, stream>>>(
        hgbuf, w2_0b, w2_1b, fc2_b0, fc2_b1, pick2, d_out, T, Dh, C);
  }
}
